// FrequencyMoE_50680614093045
// MI455X (gfx1250) — matrix operand order repaired and checked by reading, not compiled, not run on hardware
//
#include <hip/hip_runtime.h>
#include <hip/hip_bf16.h>
#include <math.h>

// ---------------- problem constants ----------------
#define BB   8
#define SS   4096
#define DD   512
#define FF_  2049                  // rfft bins
#define NT   (BB * FF_)            // 16392 tokens (b,f)
#define NEXP 8
#define DFF  2048
#define FREQD (2 * DD)             // 1024
#define NR   (2 * NT)              // 32784 expert-row assignments (top-2)

typedef __attribute__((ext_vector_type(16))) __bf16          v16bf;
typedef __attribute__((ext_vector_type(16))) unsigned short  v16us;
typedef __attribute__((ext_vector_type(8)))  unsigned short  us8;
typedef __attribute__((ext_vector_type(8)))  float           v8f;

union FragBF { v16us us; v16bf bf; };

#if defined(__gfx1250__) && __has_builtin(__builtin_amdgcn_global_load_async_to_lds_b128)
#define HAVE_ASYNC_LDS 1
#else
#define HAVE_ASYNC_LDS 0
#endif

__device__ __forceinline__ void wait_asynccnt0() {
#if defined(__gfx1250__) && __has_builtin(__builtin_amdgcn_s_wait_asynccnt)
    __builtin_amdgcn_s_wait_asynccnt(0);
#else
    asm volatile("s_wait_asynccnt 0x0" ::: "memory");
#endif
}

#if HAVE_ASYNC_LDS
// Builtin signature (from clang diagnostic): (v4i AS1*, v4i AS3*, imm offset, imm cpol),
// v4i = int __attribute__((vector_size(16))).  Build AS-typed pointers via int->ptr casts:
// global = full 64-bit VA; LDS = low 32 bits of generic LDS pointer (aperture => byte offset).
typedef __attribute__((vector_size(16)))    int  vsi4;
typedef __attribute__((address_space(1)))   vsi4 g_vsi4;
typedef __attribute__((address_space(3)))   vsi4 l_vsi4;

__device__ __forceinline__ void async_b128(const void* g, const void* l, int n16) {
    unsigned long long ga = (unsigned long long)(size_t)g;
    unsigned           lo = (unsigned)(size_t)l;
    for (int q = 0; q < n16; q++)
        __builtin_amdgcn_global_load_async_to_lds_b128(
            (g_vsi4*)(ga + (unsigned long long)q * 16),
            (l_vsi4*)(unsigned long long)(lo + (unsigned)q * 16),
            0, 0);
}
#endif

__device__ __forceinline__ unsigned short f2bf(float f) {
    unsigned u = __float_as_uint(f);
    unsigned r = (u + 0x7FFFu + ((u >> 16) & 1u)) >> 16;   // RNE
    return (unsigned short)r;
}
__device__ __forceinline__ float gelu_exact(float x) {
    return 0.5f * x * (1.0f + erff(x * 0.70710678118654752f));
}

// ---------------- forward rfft (ortho): one (b,d) column per block, bf16 out ----------------
__global__ __launch_bounds__(256) void fft_fwd_kernel(const float* __restrict__ x,
                                                      unsigned short* __restrict__ xf_bf) {
    const int b = blockIdx.x / DD;
    const int d = blockIdx.x % DD;
    __shared__ float2 buf[SS];                       // 32 KB
    for (int i = threadIdx.x; i < SS; i += 256) {
        unsigned r = __brev((unsigned)i) >> 20;      // 12-bit reverse
        buf[r] = make_float2(x[((long)b * SS + i) * DD + d], 0.f);
    }
    __syncthreads();
    for (int len = 2; len <= SS; len <<= 1) {
        const int half = len >> 1;
        const float ang = -6.283185307179586f / (float)len;
        for (int j = threadIdx.x; j < SS / 2; j += 256) {
            int blk = j / half, pos = j - blk * half;
            int i0 = blk * len + pos, i1 = i0 + half;
            float s, c; __sincosf(ang * (float)pos, &s, &c);
            float2 u = buf[i0], v = buf[i1];
            float2 w = make_float2(v.x * c - v.y * s, v.x * s + v.y * c);
            buf[i0] = make_float2(u.x + w.x, u.y + w.y);
            buf[i1] = make_float2(u.x - w.x, u.y - w.y);
        }
        __syncthreads();
    }
    const float sc = 0.015625f;                      // 1/sqrt(4096)
    for (int f = threadIdx.x; f < FF_; f += 256) {
        float2 v = buf[f];
        xf_bf[((long)b * FF_ + f) * FREQD + d]      = f2bf(v.x * sc);
        xf_bf[((long)b * FF_ + f) * FREQD + DD + d] = f2bf(v.y * sc);
    }
}

// ---------------- inverse rfft (ortho) ----------------
__global__ __launch_bounds__(256) void fft_inv_kernel(const float* __restrict__ of,
                                                      float* __restrict__ ot) {
    const int b = blockIdx.x / DD;
    const int d = blockIdx.x % DD;
    __shared__ float2 buf[SS];
    for (int i = threadIdx.x; i < SS; i += 256) {
        float re, im;
        if (i < FF_) {
            re = of[((long)b * FF_ + i) * FREQD + d];
            im = of[((long)b * FF_ + i) * FREQD + DD + d];
        } else {
            int m = SS - i;                          // hermitian extension
            re =  of[((long)b * FF_ + m) * FREQD + d];
            im = -of[((long)b * FF_ + m) * FREQD + DD + d];
        }
        unsigned r = __brev((unsigned)i) >> 20;
        buf[r] = make_float2(re, im);
    }
    __syncthreads();
    for (int len = 2; len <= SS; len <<= 1) {
        const int half = len >> 1;
        const float ang = 6.283185307179586f / (float)len;   // inverse: +angle
        for (int j = threadIdx.x; j < SS / 2; j += 256) {
            int blk = j / half, pos = j - blk * half;
            int i0 = blk * len + pos, i1 = i0 + half;
            float s, c; __sincosf(ang * (float)pos, &s, &c);
            float2 u = buf[i0], v = buf[i1];
            float2 w = make_float2(v.x * c - v.y * s, v.x * s + v.y * c);
            buf[i0] = make_float2(u.x + w.x, u.y + w.y);
            buf[i1] = make_float2(u.x - w.x, u.y - w.y);
        }
        __syncthreads();
    }
    const float sc = 0.015625f;
    for (int i = threadIdx.x; i < SS; i += 256)
        ot[((long)b * SS + i) * DD + d] = buf[i].x * sc;
}

// (K,N) f32 row-major -> (N,K) bf16 row-major (B^T), batched over z. 32x32 LDS tiles, coalesced both ways.
__global__ __launch_bounds__(256) void transpose_cast_kernel(const float* __restrict__ in,
                                                             unsigned short* __restrict__ out,
                                                             int K, int N) {
    __shared__ float tile[32][33];
    const long zoff = (long)blockIdx.z * K * N;
    const int kb = blockIdx.x * 32, nb = blockIdx.y * 32;
    const int tx = threadIdx.x & 31, ty = threadIdx.x >> 5;   // 32 x 8
    for (int r = ty; r < 32; r += 8)
        tile[r][tx] = in[zoff + (long)(kb + r) * N + (nb + tx)];
    __syncthreads();
    for (int r = ty; r < 32; r += 8)
        out[zoff + (long)(nb + r) * K + (kb + tx)] = f2bf(tile[tx][r]);
}

// ---------------- bf16 WMMA GEMM ----------------
// C[M,N] = act( A[M,K] @ Bt[N,K]^T + bias ), macro-tile 128x64, 8 waves of 32x32.
// Stage BK=64 (2 WMMA K-chunks per stage). Tile rows beyond mcount are index-clamped
// (never zero-filled): their accumulator rows are masked at the store, so loads stay
// unconditional -> clean async/unpredicated load path.
// EPI 0: f32 out + bias.  EPI 1: bf16 out + bias + exact GELU.  EPI 2: f32 + bf16 dual out.
#define LDT 72   // padded LDS row pitch (bf16 elems): 144B, 16B-aligned, bank stride 36
template <int EPI>
__global__ __launch_bounds__(256) void gemm_bf16_kernel(
    const unsigned short* __restrict__ A, const unsigned short* __restrict__ Bt,
    const float* __restrict__ bias, void* __restrict__ C, unsigned short* __restrict__ C2,
    int M, int N, int K,
    const int* __restrict__ rowlist, const int* __restrict__ counts, const int* __restrict__ offsets,
    long strideBtZ, long strideBiasZ) {

    const int tid  = threadIdx.x;
    const int wid  = tid >> 5;
    const int lane = tid & 31;
    const int z    = blockIdx.z;
    const int tileN = blockIdx.x * 64;
    const int tileM = blockIdx.y * 128;

    const int zbase  = offsets ? offsets[z] : 0;
    const int mcount = counts  ? counts[z]  : M;
    if (tileM >= mcount) return;

    const unsigned short* Btz   = Bt   + (long)z * strideBtZ;
    const float*          biasz = bias + (long)z * strideBiasZ;

    // ---- hoisted per-thread load descriptors (K-invariant, index-clamped) ----
    const int ar = tid >> 1;                 // A row in tile
    const int ac = (tid & 1) * 32;           // A col base (32 elems/stage per thread)
    const int am = tileM + ar;
    const int amc = (am < mcount) ? am : (mcount - 1);        // clamp, store-masked later
    const int arow = rowlist ? rowlist[zbase + amc] : (zbase + amc);
    const unsigned short* asrc = A + (long)arow * K + ac;

    const int br = tid >> 2;                 // B row (N dim) in tile
    const int bc = (tid & 3) * 16;           // B col base (16 elems/stage per thread)
    const unsigned short* bsrc = Btz + (long)(tileN + br) * K + bc;

    const int wm = (wid & 3) * 32;           // 4 waves along M
    const int wn = (wid >> 2) * 32;          // 2 waves along N
    const int hi  = lane >> 4;
    const int l16 = lane & 15;

    v8f acc[2][2];
    #pragma unroll
    for (int i = 0; i < 2; i++)
        #pragma unroll
        for (int j = 0; j < 2; j++) acc[i][j] = (v8f)0.f;

    // fragment math for one resident stage
    auto compute_stage = [&](const unsigned short (*Asb)[LDT], const unsigned short (*Bsb)[LDT]) {
        #pragma unroll
        for (int kk = 0; kk < 64; kk += 32) {
            FragBF a[2], b[2];
            #pragma unroll
            for (int i = 0; i < 2; i++) {    // A: lane=row(l16), K split 8+8 by hi
                const unsigned short* ap = &Asb[wm + i * 16 + l16][kk + hi * 8];
                us8 lo = *(const us8*)ap;
                us8 up = *(const us8*)(ap + 16);
                #pragma unroll
                for (int q = 0; q < 8; q++) { a[i].us[q] = lo[q]; a[i].us[q + 8] = up[q]; }
            }
            #pragma unroll
            for (int j = 0; j < 2; j++) {    // B: lane=col(l16), K split 8+8 by hi
                const unsigned short* bp = &Bsb[wn + j * 16 + l16][kk + hi * 8];
                us8 lo = *(const us8*)bp;
                us8 up = *(const us8*)(bp + 16);
                #pragma unroll
                for (int q = 0; q < 8; q++) { b[j].us[q] = lo[q]; b[j].us[q + 8] = up[q]; }
            }
            #pragma unroll
            for (int i = 0; i < 2; i++)
                #pragma unroll
                for (int j = 0; j < 2; j++)
                    acc[i][j] = __builtin_amdgcn_wmma_f32_16x16x32_bf16(
                        false, a[i].bf, false, b[j].bf, (short)0, acc[i][j], false, false);
        }
    };

#if HAVE_ASYNC_LDS
    // ---- async global->LDS DMA path, LDS double-buffered (ASYNCcnt) ----
    __shared__ unsigned short As[2][128][LDT];
    __shared__ unsigned short Bs[2][64][LDT];
    auto issue_stage = [&](int k0, int sbuf) {
        async_b128(asrc + k0, &As[sbuf][ar][ac], 4);   // 64B of A row
        async_b128(bsrc + k0, &Bs[sbuf][br][bc], 2);   // 32B of B^T row
    };
    const int nst = K >> 6;
    issue_stage(0, 0);
    wait_asynccnt0();
    __syncthreads();
    for (int s = 0; s < nst; s++) {
        const int cur = s & 1;
        if (s + 1 < nst) issue_stage((s + 1) << 6, cur ^ 1);   // DMA overlaps WMMA below
        compute_stage(As[cur], Bs[cur]);
        if (s + 1 < nst) wait_asynccnt0();
        __syncthreads();
    }
#else
    // ---- register double-buffered sync-load path ----
    __shared__ unsigned short As[128][LDT];
    __shared__ unsigned short Bs[64][LDT];
    us8 areg[4], breg[2];
    #pragma unroll
    for (int q = 0; q < 4; q++) areg[q] = *(const us8*)(asrc + q * 8);
    #pragma unroll
    for (int q = 0; q < 2; q++) breg[q] = *(const us8*)(bsrc + q * 8);

    for (int k0 = 0; k0 < K; k0 += 64) {
        #pragma unroll
        for (int q = 0; q < 4; q++) *(us8*)&As[ar][ac + q * 8] = areg[q];
        #pragma unroll
        for (int q = 0; q < 2; q++) *(us8*)&Bs[br][bc + q * 8] = breg[q];
        __syncthreads();
        if (k0 + 64 < K) {
            #pragma unroll
            for (int q = 0; q < 4; q++) areg[q] = *(const us8*)(asrc + k0 + 64 + q * 8);
            #pragma unroll
            for (int q = 0; q < 2; q++) breg[q] = *(const us8*)(bsrc + k0 + 64 + q * 8);
        }
        compute_stage(As, Bs);
        __syncthreads();
    }
#endif

    // epilogue: VGPR r -> (m = 16i + 8*hi + r, n = 16j + l16)
    #pragma unroll
    for (int i = 0; i < 2; i++) {
        #pragma unroll
        for (int j = 0; j < 2; j++) {
            int nbase = tileN + wn + j * 16 + l16;
            float bv = biasz[nbase];
            #pragma unroll
            for (int r = 0; r < 8; r++) {
                int m = tileM + wm + i * 16 + hi * 8 + r;
                if (m < mcount) {
                    float v = acc[i][j][r] + bv;
                    long idx = (long)(zbase + m) * N + nbase;
                    if (EPI == 1) {
                        C2[idx] = f2bf(gelu_exact(v));
                    } else if (EPI == 2) {
                        ((float*)C)[idx] = v;
                        C2[idx] = f2bf(v);
                    } else {
                        ((float*)C)[idx] = v;
                    }
                }
            }
        }
    }
}

// ---------------- router + top-2 softmax + expert counting ----------------
__global__ __launch_bounds__(256) void router_topk_kernel(
    const float* __restrict__ h, const float* __restrict__ rw, const float* __restrict__ rb,
    int* __restrict__ tk_idx, float* __restrict__ tk_w, int* __restrict__ counts) {
    const int t = blockIdx.x;
    __shared__ float red[NEXP][256];
    const float* hr = h + (long)t * DD;
    float p[NEXP];
    for (int e = 0; e < NEXP; e++) p[e] = 0.f;
    for (int d = threadIdx.x; d < DD; d += 256) {
        float hv = hr[d];
        for (int e = 0; e < NEXP; e++) p[e] += hv * rw[d * NEXP + e];
    }
    for (int e = 0; e < NEXP; e++) red[e][threadIdx.x] = p[e];
    __syncthreads();
    for (int s = 128; s > 0; s >>= 1) {
        if (threadIdx.x < s)
            for (int e = 0; e < NEXP; e++) red[e][threadIdx.x] += red[e][threadIdx.x + s];
        __syncthreads();
    }
    if (threadIdx.x == 0) {
        float lg[NEXP];
        for (int e = 0; e < NEXP; e++) lg[e] = red[e][0] + rb[e];
        int i0 = 0;
        for (int e = 1; e < NEXP; e++) if (lg[e] > lg[i0]) i0 = e;
        int i1 = (i0 == 0) ? 1 : 0;
        for (int e = 0; e < NEXP; e++) if (e != i0 && lg[e] > lg[i1]) i1 = e;
        float m = lg[i0];
        float e0 = __expf(lg[i0] - m), e1 = __expf(lg[i1] - m);
        float inv = 1.f / (e0 + e1);
        tk_idx[t * 2] = i0;  tk_idx[t * 2 + 1] = i1;
        tk_w[t * 2] = e0 * inv;  tk_w[t * 2 + 1] = e1 * inv;
        atomicAdd(&counts[i0], 1);
        atomicAdd(&counts[i1], 1);
    }
}

__global__ void scan8_kernel(const int* __restrict__ counts, int* __restrict__ offsets,
                             int* __restrict__ fillptr) {
    if (threadIdx.x == 0) {
        int s = 0;
        for (int e = 0; e < NEXP; e++) { offsets[e] = s; fillptr[e] = s; s += counts[e]; }
    }
}

__global__ void fill_rows_kernel(const int* __restrict__ tk_idx, int* __restrict__ fillptr,
                                 int* __restrict__ rowlist, int* __restrict__ rowidx) {
    int t = blockIdx.x * 256 + threadIdx.x;
    if (t >= NT) return;
    for (int k = 0; k < 2; k++) {
        int e = tk_idx[t * 2 + k];
        int r = atomicAdd(&fillptr[e], 1);
        rowlist[r] = t;
        rowidx[t * 2 + k] = r;
    }
}

// moe_out[t] = w0*O[r0] + w1*O[r1]  -> bf16 (consumed only by pout GEMM)
__global__ void combine_kernel(const float* __restrict__ O, const int* __restrict__ rowidx,
                               const float* __restrict__ tk_w, unsigned short* __restrict__ moe_bf) {
    long id = (long)blockIdx.x * 256 + threadIdx.x;
    if (id >= (long)NT * DD) return;
    int t = (int)(id >> 9);
    int d = (int)(id & (DD - 1));
    int r0 = rowidx[t * 2], r1 = rowidx[t * 2 + 1];
    float v = tk_w[t * 2] * O[(long)r0 * DD + d] + tk_w[t * 2 + 1] * O[(long)r1 * DD + d];
    moe_bf[id] = f2bf(v);
}

// ---------------- layernorm + residual ----------------
__global__ __launch_bounds__(256) void ln_res_kernel(
    const float* __restrict__ ot, const float* __restrict__ x,
    const float* __restrict__ g, const float* __restrict__ be, float* __restrict__ out) {
    const long row = blockIdx.x;
    const float* r = ot + row * DD;
    __shared__ float red[256];
    float s = 0.f;
    for (int d = threadIdx.x; d < DD; d += 256) s += r[d];
    red[threadIdx.x] = s; __syncthreads();
    for (int k = 128; k > 0; k >>= 1) {
        if (threadIdx.x < k) red[threadIdx.x] += red[threadIdx.x + k];
        __syncthreads();
    }
    float mu = red[0] * (1.f / DD);
    __syncthreads();
    float v = 0.f;
    for (int d = threadIdx.x; d < DD; d += 256) { float t = r[d] - mu; v += t * t; }
    red[threadIdx.x] = v; __syncthreads();
    for (int k = 128; k > 0; k >>= 1) {
        if (threadIdx.x < k) red[threadIdx.x] += red[threadIdx.x + k];
        __syncthreads();
    }
    float rstd = rsqrtf(red[0] * (1.f / DD) + 1e-5f);
    for (int d = threadIdx.x; d < DD; d += 256)
        out[row * DD + d] = (r[d] - mu) * rstd * g[d] + be[d] + x[row * DD + d];
}

// ---------------- host launch ----------------
extern "C" void kernel_launch(void* const* d_in, const int* in_sizes, int n_in,
                              void* d_out, int out_size, void* d_ws, size_t ws_size,
                              hipStream_t stream) {
    const float* x        = (const float*)d_in[0];
    const float* pin_w    = (const float*)d_in[1];
    const float* pin_b    = (const float*)d_in[2];
    const float* router_w = (const float*)d_in[3];
    const float* router_b = (const float*)d_in[4];
    const float* w1       = (const float*)d_in[5];
    const float* b1       = (const float*)d_in[6];
    const float* w2       = (const float*)d_in[7];
    const float* b2       = (const float*)d_in[8];
    const float* pout_w   = (const float*)d_in[9];
    const float* pout_b   = (const float*)d_in[10];
    const float* ln_g     = (const float*)d_in[11];
    const float* ln_b     = (const float*)d_in[12];
    float* out = (float*)d_out;

    // workspace carve-up (256B aligned)
    char* p = (char*)d_ws;
    auto alloc = [&](size_t bytes) -> void* {
        void* r = (void*)p;
        p += (bytes + 255) & ~(size_t)255;
        return r;
    };
    unsigned short* xf_bf  = (unsigned short*)alloc((size_t)NT * FREQD * 2);
    float*          h_f32  = (float*)alloc((size_t)NT * DD * 4);
    unsigned short* h_bf   = (unsigned short*)alloc((size_t)NT * DD * 2);
    unsigned short* pinT   = (unsigned short*)alloc((size_t)FREQD * DD * 2);
    unsigned short* poutT  = (unsigned short*)alloc((size_t)DD * FREQD * 2);
    unsigned short* w1T    = (unsigned short*)alloc((size_t)NEXP * DFF * DD * 2);
    unsigned short* w2T    = (unsigned short*)alloc((size_t)NEXP * DD * DFF * 2);
    int*            tk_idx = (int*)alloc((size_t)NT * 2 * 4);
    float*          tk_w   = (float*)alloc((size_t)NT * 2 * 4);
    int*            counts = (int*)alloc(8 * 4);
    int*            offs   = (int*)alloc(8 * 4);
    int*            fillp  = (int*)alloc(8 * 4);
    int*            rowlist= (int*)alloc((size_t)NR * 4);
    int*            rowidx = (int*)alloc((size_t)NT * 2 * 4);
    unsigned short* hid_bf = (unsigned short*)alloc((size_t)NR * DFF * 2);
    float*          O_f32  = (float*)alloc((size_t)NR * DD * 4);
    unsigned short* moe_bf = (unsigned short*)alloc((size_t)NT * DD * 2);
    float*          of     = (float*)alloc((size_t)NT * FREQD * 4);
    float*          ot     = (float*)alloc((size_t)BB * SS * DD * 4);

    (void)hipMemsetAsync(counts, 0, 8 * sizeof(int), stream);

    // 1) forward rfft -> xf_bf (ortho, bf16)
    fft_fwd_kernel<<<BB * DD, 256, 0, stream>>>(x, xf_bf);

    // 2) weight prep: bf16 transposed copies (tiled, coalesced)
    transpose_cast_kernel<<<dim3(FREQD / 32, DD / 32, 1), 256, 0, stream>>>(pin_w, pinT, FREQD, DD);
    transpose_cast_kernel<<<dim3(DD / 32, FREQD / 32, 1), 256, 0, stream>>>(pout_w, poutT, DD, FREQD);
    transpose_cast_kernel<<<dim3(DD / 32, DFF / 32, NEXP), 256, 0, stream>>>(w1, w1T, DD, DFF);
    transpose_cast_kernel<<<dim3(DFF / 32, DD / 32, NEXP), 256, 0, stream>>>(w2, w2T, DFF, DD);

    // 3) pin GEMM: h = xf @ pin_w + pin_b  (dual f32 + bf16 out)
    gemm_bf16_kernel<2><<<dim3(DD / 64, (NT + 127) / 128, 1), 256, 0, stream>>>(
        xf_bf, pinT, pin_b, h_f32, h_bf, NT, DD, FREQD, nullptr, nullptr, nullptr, 0, 0);

    // 4) routing: logits, top-2 softmax, per-expert counts, row lists
    router_topk_kernel<<<NT, 256, 0, stream>>>(h_f32, router_w, router_b, tk_idx, tk_w, counts);
    scan8_kernel<<<1, 32, 0, stream>>>(counts, offs, fillp);
    fill_rows_kernel<<<(NT + 255) / 256, 256, 0, stream>>>(tk_idx, fillp, rowlist, rowidx);

    // 5) expert GEMM1 (gathered): hid = gelu(h @ w1[e] + b1[e]) -> bf16
    gemm_bf16_kernel<1><<<dim3(DFF / 64, (NR + 127) / 128, NEXP), 256, 0, stream>>>(
        h_bf, w1T, b1, nullptr, hid_bf, NR, DFF, DD, rowlist, counts, offs,
        (long)DFF * DD, (long)DFF);

    // 6) expert GEMM2: O = hid @ w2[e] + b2[e]
    gemm_bf16_kernel<0><<<dim3(DD / 64, (NR + 127) / 128, NEXP), 256, 0, stream>>>(
        hid_bf, w2T, b2, O_f32, nullptr, NR, DD, DFF, nullptr, counts, offs,
        (long)DD * DFF, (long)DD);

    // 7) weighted top-2 combine -> bf16
    combine_kernel<<<(int)(((long)NT * DD + 255) / 256), 256, 0, stream>>>(O_f32, rowidx, tk_w, moe_bf);

    // 8) pout GEMM: of = moe @ pout_w + pout_b
    gemm_bf16_kernel<0><<<dim3(FREQD / 64, (NT + 127) / 128, 1), 256, 0, stream>>>(
        moe_bf, poutT, pout_b, of, nullptr, NT, FREQD, DD, nullptr, nullptr, nullptr, 0, 0);

    // 9) inverse rfft -> time domain
    fft_inv_kernel<<<BB * DD, 256, 0, stream>>>(of, ot);

    // 10) layernorm + residual -> out
    ln_res_kernel<<<BB * SS, 256, 0, stream>>>(ot, x, ln_g, ln_b, out);
}
